// CSIPoseEstimationModel_79001628442859
// MI455X (gfx1250) — compile-verified, matrix operand order checked by reading, not run on hardware
//
#include <hip/hip_runtime.h>
#include <math.h>

// ---------------------------------------------------------------------------
// CSI pose-estimation model for MI455X (gfx1250, wave32, WMMA).
// All dense channel contractions (pointwise conv1d, 1x1/1x3 conv2d via
// im2col, qkv projection, final head) run through v_wmma_f32_16x16x32_f16
// on pre-packed, zero-padded f16 operands: the inner loop is 4 x b128 loads
// + 1 WMMA, no branches. Axial attention is fused (two-pass, no LxL tensor).
// ---------------------------------------------------------------------------

typedef __attribute__((ext_vector_type(16))) _Float16 v16h;
typedef __attribute__((ext_vector_type(8)))  _Float16 v8h;
typedef __attribute__((ext_vector_type(8)))  float    v8f;

#define EPSV 1e-5f

// ============================ WMMA GEMM core ===============================
// Y[nb, m, l] = sum_k Wp[m,k] * Xp[nb, l, k]  (+ bias[m] if bias != null)
// Wp: (Mpad, Kpad) f16, zero-padded.  Xp: (NB, L, Kpad) f16, zero-padded.
// Kpad % 32 == 0, L % 16 == 0. One 16x16 tile per wave (wave-uniform exit,
// EXEC all-ones across WMMA per ISA 7.12).
__global__ __launch_bounds__(256)
void k_gemm_wmma(const _Float16* __restrict__ Wp, const _Float16* __restrict__ Xp,
                 const float* __restrict__ bias, float* __restrict__ Y,
                 int M, int Kpad, int L, int NB) {
  const int lane   = threadIdx.x & 31;
  const int wave   = threadIdx.x >> 5;
  const long N     = (long)NB * L;
  const int tilesN = (int)(N >> 4);
  const int tilesM = (M + 15) >> 4;
  long tile = (long)blockIdx.x * (blockDim.x >> 5) + wave;
  if (tile >= (long)tilesM * tilesN) return;   // wave-uniform exit
  const int tm = (int)(tile / tilesN);
  const int tn = (int)(tile % tilesN);
  const int m0 = tm << 4;
  const int n0 = tn << 4;
  const int nb = n0 / L;          // tile never crosses a batch (L % 16 == 0)
  const int l0 = n0 % L;
  const int half   = lane >> 4;   // 0 | 1
  const int lanelo = lane & 15;
  const int l      = l0 + lanelo;

  // Fragment layout (16-bit A 16x32): lane's data = two contiguous 8-half
  // chunks at K offsets half*8 and 16+half*8. Same pattern for B with the
  // K-major Xp layout.
  const _Float16* __restrict__ arow = Wp + (long)(m0 + lanelo) * Kpad + half * 8;
  const _Float16* __restrict__ brow = Xp + ((long)nb * L + l) * Kpad + half * 8;

  v8f acc = {};
  for (int k0 = 0; k0 < Kpad; k0 += 32) {
    __builtin_prefetch(arow + k0 + 64, 0, 1);
    __builtin_prefetch(brow + k0 + 64, 0, 1);
    const v8h alo = *(const v8h*)(arow + k0);
    const v8h ahi = *(const v8h*)(arow + k0 + 16);
    const v8h blo = *(const v8h*)(brow + k0);
    const v8h bhi = *(const v8h*)(brow + k0 + 16);
    const v16h a = __builtin_shufflevector(alo, ahi, 0, 1, 2, 3, 4, 5, 6, 7,
                                           8, 9, 10, 11, 12, 13, 14, 15);
    const v16h b = __builtin_shufflevector(blo, bhi, 0, 1, 2, 3, 4, 5, 6, 7,
                                           8, 9, 10, 11, 12, 13, 14, 15);
    acc = __builtin_amdgcn_wmma_f32_16x16x32_f16(
        /*neg_a=*/false, a, /*neg_b=*/false, b,
        /*c_mod=*/(short)0, acc, /*reuse_a=*/false, /*reuse_b=*/false);
  }
  // C/D layout: VGPR i -> M = m0 + i + 8*half, N = n0 + lanelo
#pragma unroll
  for (int i = 0; i < 8; ++i) {
    const int m = m0 + i + 8 * half;
    if (m < M) {
      float v = acc[i];
      if (bias) v += bias[m];
      Y[((long)nb * M + m) * L + l] = v;
    }
  }
}

// =========================== operand packing ===============================
// Weights (M, K) f32 -> (Mpad, Kpad) f16 zero-padded.
__global__ __launch_bounds__(256)
void k_pack_w(const float* __restrict__ W, _Float16* __restrict__ Wp,
              int M, int K, int Kpad, long total) {
  long idx = (long)blockIdx.x * blockDim.x + threadIdx.x;
  if (idx >= total) return;
  const int k = (int)(idx % Kpad);
  const long m = idx / Kpad;
  Wp[idx] = (m < M && k < K) ? (_Float16)W[m * K + k] : (_Float16)0.f;
}

// Activations (NB, K, L) f32 -> K-major (NB, L, Kpad) f16 zero-padded.
__global__ __launch_bounds__(256)
void k_pack_xT(const float* __restrict__ X, _Float16* __restrict__ Xp,
               int K, int L, int Kpad, long total) {
  long idx = (long)blockIdx.x * blockDim.x + threadIdx.x;
  if (idx >= total) return;
  const int k = (int)(idx % Kpad);
  long t = idx / Kpad;
  const int l = (int)(t % L);
  const long nb = t / L;
  Xp[idx] = (k < K) ? (_Float16)X[(nb * K + k) * L + l] : (_Float16)0.f;
}

// im2col for 1x3 conv along W (pad 1, stride s): (B,Cin,H,Win) f32 ->
// (B, H*Wout, Kpad) f16 with K = Cin*3 (k = ci*3 + tap), zero-padded.
__global__ __launch_bounds__(256)
void k_im2col_1x3(const float* __restrict__ x, _Float16* __restrict__ Xp,
                  int Cin, int H, int Win, int Wout, int stride, int Kpad,
                  long total) {
  long idx = (long)blockIdx.x * blockDim.x + threadIdx.x;
  if (idx >= total) return;
  const int k = (int)(idx % Kpad);
  long t = idx / Kpad;
  const int wo = (int)(t % Wout);  t /= Wout;
  const int h  = (int)(t % H);
  const long b = t / H;
  float v = 0.f;
  if (k < Cin * 3) {
    const int ci  = k / 3;
    const int tap = k - ci * 3;
    const int wi  = wo * stride + tap - 1;
    if (wi >= 0 && wi < Win)
      v = x[((b * Cin + ci) * H + h) * (long)Win + wi];
  }
  Xp[idx] = (_Float16)v;
}

// im2col for 1x1 conv with W-stride: K = Cin.
__global__ __launch_bounds__(256)
void k_im2col_1x1(const float* __restrict__ x, _Float16* __restrict__ Xp,
                  int Cin, int H, int Win, int Wout, int stride, int Kpad,
                  long total) {
  long idx = (long)blockIdx.x * blockDim.x + threadIdx.x;
  if (idx >= total) return;
  const int k = (int)(idx % Kpad);
  long t = idx / Kpad;
  const int wo = (int)(t % Wout);  t /= Wout;
  const int h  = (int)(t % H);
  const long b = t / H;
  float v = 0.f;
  if (k < Cin)
    v = x[((b * Cin + k) * H + h) * (long)Win + wo * stride];
  Xp[idx] = (_Float16)v;
}

// ======================= BatchNorm (training stats) ========================
__global__ __launch_bounds__(256)
void k_bn_stats(const float* __restrict__ x, float* __restrict__ mean,
                float* __restrict__ istd, int C, int NB, long S) {
  const int c = blockIdx.x;
  const long cnt = (long)NB * S;
  float s = 0.f, q = 0.f;
  for (long i = threadIdx.x; i < cnt; i += blockDim.x) {
    long b = i / S, sp = i - b * S;
    float v = x[(b * C + c) * S + sp];
    s += v; q += v * v;
  }
#pragma unroll
  for (int o = 16; o > 0; o >>= 1) {
    s += __shfl_xor(s, o, 32);
    q += __shfl_xor(q, o, 32);
  }
  __shared__ float shs[8], shq[8];
  const int wv = threadIdx.x >> 5, lane = threadIdx.x & 31;
  if (lane == 0) { shs[wv] = s; shq[wv] = q; }
  __syncthreads();
  if (threadIdx.x == 0) {
    float ts = 0.f, tq = 0.f;
    const int nw = blockDim.x >> 5;
    for (int w = 0; w < nw; ++w) { ts += shs[w]; tq += shq[w]; }
    float m = ts / (float)cnt;
    float v = tq / (float)cnt - m * m;
    mean[c] = m;
    istd[c] = rsqrtf(v + EPSV);
  }
}

__global__ __launch_bounds__(256)
void k_bn_act(const float* __restrict__ x, float* __restrict__ y,
              const float* __restrict__ mean, const float* __restrict__ istd,
              const float* __restrict__ g, const float* __restrict__ b,
              int C, long S, long total, int act) {
  long i = (long)blockIdx.x * blockDim.x + threadIdx.x;
  if (i >= total) return;
  int c = (int)((i / S) % C);
  float v = (x[i] - mean[c]) * istd[c] * g[c] + b[c];
  if (act) v = v / (1.f + expf(-v));
  y[i] = v;
}

__global__ __launch_bounds__(256)
void k_add_silu(float* __restrict__ y, const float* __restrict__ a,
                const float* __restrict__ b, long total) {
  long i = (long)blockIdx.x * blockDim.x + threadIdx.x;
  if (i >= total) return;
  float v = a[i] + b[i];
  y[i] = v / (1.f + expf(-v));
}

__global__ void k_zero(float* __restrict__ p, int n) {
  int i = threadIdx.x;
  if (i < n) p[i] = 0.f;
}

// ================= grouped dilated causal conv1d (k=3) =====================
__global__ __launch_bounds__(256)
void k_gconv1d(const float* __restrict__ x, const float* __restrict__ w,
               float* __restrict__ y, int B, int Cin, int Cout, int L,
               int dil, int groups) {
  long idx = (long)blockIdx.x * blockDim.x + threadIdx.x;
  long total = (long)B * Cout * L;
  if (idx >= total) return;
  const int l  = (int)(idx % L);
  const int co = (int)((idx / L) % Cout);
  const int b  = (int)(idx / ((long)L * Cout));
  const int cing  = Cin / groups;
  const int coutg = Cout / groups;
  const int gi    = co / coutg;
  const float* __restrict__ xb = x + ((long)b * Cin + gi * cing) * L;
  const float* __restrict__ wb = w + (long)co * cing * 3;
  float acc = 0.f;
  for (int ci = 0; ci < cing; ++ci) {
#pragma unroll
    for (int t = 0; t < 3; ++t) {
      const int li = l + (t - 2) * dil;   // left pad = 2*dil (causal)
      if (li >= 0) acc += wb[ci * 3 + t] * xb[(long)ci * L + li];
    }
  }
  y[idx] = acc;
}

// ====================== generic 4D gather-permute ==========================
__global__ __launch_bounds__(256)
void k_permute4(const float* __restrict__ x, float* __restrict__ y, long total,
                int d1, int d2, int d3, long s0, long s1, long s2, long s3) {
  long idx = (long)blockIdx.x * blockDim.x + threadIdx.x;
  if (idx >= total) return;
  const int i3 = (int)(idx % d3);  long t = idx / d3;
  const int i2 = (int)(t % d2);    t /= d2;
  const int i1 = (int)(t % d1);
  const long i0 = t / d1;
  y[idx] = x[i0 * s0 + i1 * s1 + (long)i2 * s2 + (long)i3 * s3];
}

// ================== fused axial attention (two passes) =====================
// qkv layout: (NB, 192, L); groups=8, gp=8.
__global__ __launch_bounds__(256)
void k_attn_stats(const float* __restrict__ qkv, float* __restrict__ sums,
                  int NB, int L) {
  __shared__ float ls[16];
  if (threadIdx.x < 16) ls[threadIdx.x] = 0.f;
  __syncthreads();
  const long r = (long)blockIdx.x * blockDim.x + threadIdx.x;
  const long total = (long)NB * 8 * L;
  if (r < total) {
    const int i = (int)(r % L);
    const int g = (int)((r / L) % 8);
    const long nb = r / ((long)L * 8);
    const float* __restrict__ base = qkv + nb * (long)192 * L;
    const float* __restrict__ Q  = base + (long)(g * 8) * L;
    const float* __restrict__ Kk = base + (long)(64 + g * 8) * L;
    float qv[8];
#pragma unroll
    for (int c = 0; c < 8; ++c) qv[c] = Q[(long)c * L + i];
    float s = 0.f, q = 0.f;
    for (int j = 0; j < L; ++j) {
      float t = 0.f;
#pragma unroll
      for (int c = 0; c < 8; ++c) t += qv[c] * Kk[(long)c * L + j];
      s += t; q += t * t;
    }
    atomicAdd(&ls[g], s);
    atomicAdd(&ls[8 + g], q);
  }
  __syncthreads();
  if (threadIdx.x < 16) atomicAdd(&sums[threadIdx.x], ls[threadIdx.x]);
}

__global__ void k_stats_finalize(const float* __restrict__ sums,
                                 float* __restrict__ mean, float* __restrict__ istd,
                                 int n, float cnt) {
  int i = threadIdx.x;
  if (i < n) {
    float m = sums[i] / cnt;
    float v = sums[n + i] / cnt - m * m;
    mean[i] = m;
    istd[i] = rsqrtf(v + EPSV);
  }
}

// One wave per output row: recompute qk row, BN, wave32 softmax via
// __shfl_xor, contract with V. sv layout: (NB, 64, L).
__global__ __launch_bounds__(256)
void k_attn_apply(const float* __restrict__ qkv, const float* __restrict__ mean,
                  const float* __restrict__ istd, const float* __restrict__ gs,
                  const float* __restrict__ gb, float* __restrict__ sv,
                  int NB, int L) {
  const int lane = threadIdx.x & 31, wv = threadIdx.x >> 5;
  const long row = (long)blockIdx.x * (blockDim.x >> 5) + wv;
  const long total = (long)NB * 8 * L;
  if (row >= total) return;               // wave-uniform
  const int i = (int)(row % L);
  const int g = (int)((row / L) % 8);
  const long nb = row / ((long)L * 8);
  const float* __restrict__ base = qkv + nb * (long)192 * L;
  const float* __restrict__ Q  = base + (long)(g * 8) * L;
  const float* __restrict__ Kk = base + (long)(64 + g * 8) * L;
  const float* __restrict__ V  = base + (long)(128 + g * 8) * L;
  float qv[8];
#pragma unroll
  for (int c = 0; c < 8; ++c) qv[c] = Q[(long)c * L + i];
  const float m_ = mean[g], is_ = istd[g], gg = gs[g], bb = gb[g];
  float e[9];                              // L <= 272 -> <= 9 per lane
  float mx = -3.0e38f;
  int n = 0;
  for (int j = lane; j < L; j += 32, ++n) {
    float t = 0.f;
#pragma unroll
    for (int c = 0; c < 8; ++c) t += qv[c] * Kk[(long)c * L + j];
    t = (t - m_) * is_ * gg + bb;
    e[n] = t;
    mx = fmaxf(mx, t);
  }
#pragma unroll
  for (int o = 16; o > 0; o >>= 1) mx = fmaxf(mx, __shfl_xor(mx, o, 32));
  float se = 0.f;
  n = 0;
  for (int j = lane; j < L; j += 32, ++n) { e[n] = expf(e[n] - mx); se += e[n]; }
#pragma unroll
  for (int o = 16; o > 0; o >>= 1) se += __shfl_xor(se, o, 32);
  const float inv = 1.f / se;
  float acc[8] = {0.f, 0.f, 0.f, 0.f, 0.f, 0.f, 0.f, 0.f};
  n = 0;
  for (int j = lane; j < L; j += 32, ++n) {
    const float wj = e[n] * inv;
#pragma unroll
    for (int c = 0; c < 8; ++c) acc[c] += wj * V[(long)c * L + j];
  }
#pragma unroll
  for (int c = 0; c < 8; ++c)
#pragma unroll
    for (int o = 16; o > 0; o >>= 1) acc[c] += __shfl_xor(acc[c], o, 32);
  if (lane == 0) {
#pragma unroll
    for (int c = 0; c < 8; ++c)
      sv[(nb * 64 + g * 8 + c) * (long)L + i] = acc[c];
  }
}

// =========================== host orchestration ============================
namespace {

struct Ws {
  char* base; size_t used, cap;
  float* alloc(long n) {
    size_t bytes = (((size_t)n * 4) + 255) & ~(size_t)255;
    float* p = (float*)(base + used);
    used += bytes;
    return p;
  }
  _Float16* alloch(long n) {
    size_t bytes = (((size_t)n * 2) + 255) & ~(size_t)255;
    _Float16* p = (_Float16*)(base + used);
    used += bytes;
    return p;
  }
};

struct Cur {
  void* const* in; int i;
  const float* next() { return (const float*)in[i++]; }
};

inline int nblk(long total, int tpb) { return (int)((total + tpb - 1) / tpb); }

void gemm_core(hipStream_t s, const _Float16* Wp, const _Float16* Xp,
               const float* bias, float* Y, int M, int Kpad, int L, int NB) {
  long tiles = (long)((M + 15) / 16) * (((long)NB * L) / 16);
  k_gemm_wmma<<<nblk(tiles, 8), 256, 0, s>>>(Wp, Xp, bias, Y, M, Kpad, L, NB);
}

_Float16* pack_w(Ws& ws, hipStream_t s, const float* W, int M, int K, int& Kpad) {
  const int Mpad = (M + 15) & ~15;
  Kpad = (K + 31) & ~31;
  long total = (long)Mpad * Kpad;
  _Float16* Wp = ws.alloch(total);
  k_pack_w<<<nblk(total, 256), 256, 0, s>>>(W, Wp, M, K, Kpad, total);
  return Wp;
}

// Y (NB,M,L) = W (M,K) * X (NB,K,L) [+bias]
void gemm(Ws& ws, hipStream_t s, const float* W, const float* X,
          const float* bias, float* Y, int M, int K, int L, int NB) {
  size_t mark = ws.used;
  int Kpad;
  _Float16* Wp = pack_w(ws, s, W, M, K, Kpad);
  long total = (long)NB * L * Kpad;
  _Float16* Xp = ws.alloch(total);
  k_pack_xT<<<nblk(total, 256), 256, 0, s>>>(X, Xp, K, L, Kpad, total);
  gemm_core(s, Wp, Xp, bias, Y, M, Kpad, L, NB);
  ws.used = mark;
}

// y (B,Cout,H,Wout) = conv2d_1x3(x (B,Cin,H,Win)) via im2col + WMMA GEMM
void conv1x3(Ws& ws, hipStream_t s, const float* x, const float* w,
             const float* bias, float* y, int B, int Cin, int Cout,
             int H, int Win, int Wout, int stride) {
  size_t mark = ws.used;
  int Kpad;
  _Float16* Wp = pack_w(ws, s, w, Cout, Cin * 3, Kpad);
  const int L = H * Wout;
  long total = (long)B * L * Kpad;
  _Float16* Xp = ws.alloch(total);
  k_im2col_1x3<<<nblk(total, 256), 256, 0, s>>>(x, Xp, Cin, H, Win, Wout,
                                                stride, Kpad, total);
  gemm_core(s, Wp, Xp, bias, y, Cout, Kpad, L, B);
  ws.used = mark;
}

void conv1x1(Ws& ws, hipStream_t s, const float* x, const float* w,
             const float* bias, float* y, int B, int Cin, int Cout,
             int H, int Win, int Wout, int stride) {
  size_t mark = ws.used;
  int Kpad;
  _Float16* Wp = pack_w(ws, s, w, Cout, Cin, Kpad);
  const int L = H * Wout;
  long total = (long)B * L * Kpad;
  _Float16* Xp = ws.alloch(total);
  k_im2col_1x1<<<nblk(total, 256), 256, 0, s>>>(x, Xp, Cin, H, Win, Wout,
                                                stride, Kpad, total);
  gemm_core(s, Wp, Xp, bias, y, Cout, Kpad, L, B);
  ws.used = mark;
}

void bn(hipStream_t s, float* x, const float* g, const float* b,
        float* mean, float* istd, int C, int NB, long S, int act) {
  k_bn_stats<<<C, 256, 0, s>>>(x, mean, istd, C, NB, S);
  long total = (long)NB * C * S;
  k_bn_act<<<nblk(total, 256), 256, 0, s>>>(x, x, mean, istd, g, b, C, S, total, act);
}

float* tcn_block(Ws& ws, Cur& P, hipStream_t s, float* mean, float* istd,
                 const float* x, int B, int Cin, int Cout, int L, int dil) {
  const float* cg1 = P.next(); const float* g1g = P.next(); const float* g1b = P.next();
  const float* pw1 = P.next(); const float* p1g = P.next(); const float* p1b = P.next();
  const float* cg2 = P.next(); const float* g2g = P.next(); const float* g2b = P.next();
  const float* pw2 = P.next(); const float* p2g = P.next(); const float* p2b = P.next();
  const float* dsw = nullptr; const float* dsg = nullptr; const float* dsb = nullptr;
  if (Cin != Cout) { dsw = P.next(); dsg = P.next(); dsb = P.next(); }

  const long nIn = (long)B * Cin * L, nOut = (long)B * Cout * L;
  float* out = ws.alloc(nOut);
  size_t mark = ws.used;

  const float* res = x;
  if (dsw) {
    float* r = ws.alloc(nOut);
    gemm(ws, s, dsw, x, nullptr, r, Cout, Cin, L, B);
    bn(s, r, dsg, dsb, mean, istd, Cout, B, L, 0);
    res = r;
  }
  float* t1 = ws.alloc(nIn);
  k_gconv1d<<<nblk(nIn, 256), 256, 0, s>>>(x, cg1, t1, B, Cin, Cin, L, dil, 18);
  bn(s, t1, g1g, g1b, mean, istd, Cin, B, L, 1);
  float* t2 = ws.alloc(nOut);
  gemm(ws, s, pw1, t1, nullptr, t2, Cout, Cin, L, B);
  bn(s, t2, p1g, p1b, mean, istd, Cout, B, L, 1);
  float* t3 = ws.alloc(nOut);
  k_gconv1d<<<nblk(nOut, 256), 256, 0, s>>>(t2, cg2, t3, B, Cout, Cout, L, dil, 18);
  bn(s, t3, g2g, g2b, mean, istd, Cout, B, L, 1);
  float* t4 = ws.alloc(nOut);
  gemm(ws, s, pw2, t3, nullptr, t4, Cout, Cout, L, B);
  bn(s, t4, p2g, p2b, mean, istd, Cout, B, L, 1);
  k_add_silu<<<nblk(nOut, 256), 256, 0, s>>>(out, t4, res, nOut);
  ws.used = mark;
  return out;
}

float* conv_block(Ws& ws, Cur& P, hipStream_t s, float* mean, float* istd,
                  const float* x, int B, int Cin, int Cout, int H, int Win, int stride) {
  const float* c1w = P.next(); const float* c1b = P.next();
  const float* b1g = P.next(); const float* b1b = P.next();
  const float* c2w = P.next(); const float* c2b = P.next();
  const float* b2g = P.next(); const float* b2b = P.next();
  const float* c3w = P.next(); const float* c3b = P.next();
  const float* b3g = P.next(); const float* b3b = P.next();
  const float* dw  = P.next(); const float* dg  = P.next(); const float* db = P.next();

  const int Wout = (stride == 1) ? Win : (Win - 1) / stride + 1;
  const long S = (long)H * Wout;
  const long n = (long)B * Cout * S;
  float* out = ws.alloc(n);
  size_t mark = ws.used;
  float* idn = ws.alloc(n);
  float* o1  = ws.alloc(n);
  float* o2  = ws.alloc(n);
  float* o3  = ws.alloc(n);

  conv1x1(ws, s, x, dw, nullptr, idn, B, Cin, Cout, H, Win, Wout, stride);
  bn(s, idn, dg, db, mean, istd, Cout, B, S, 0);
  conv1x3(ws, s, x, c1w, c1b, o1, B, Cin, Cout, H, Win, Wout, stride);
  bn(s, o1, b1g, b1b, mean, istd, Cout, B, S, 1);
  conv1x3(ws, s, o1, c2w, c2b, o2, B, Cout, Cout, H, Wout, Wout, 1);
  bn(s, o2, b2g, b2b, mean, istd, Cout, B, S, 1);
  conv1x3(ws, s, o2, c3w, c3b, o3, B, Cout, Cout, H, Wout, Wout, 1);
  bn(s, o3, b3g, b3b, mean, istd, Cout, B, S, 0);
  k_add_silu<<<nblk(n, 256), 256, 0, s>>>(out, o3, idn, n);
  ws.used = mark;
  return out;
}

float* axial(Ws& ws, Cur& P, hipStream_t s, float* mean, float* istd, float* asums,
             const float* x, int B, int H, int W, bool width) {
  const float* qkv_w = P.next();
  const float* bqg = P.next(); const float* bqb = P.next();
  const float* bsg = P.next(); const float* bsb = P.next();
  const float* bog = P.next(); const float* bob = P.next();

  const int C = 64;
  const int NB = width ? B * H : B * W;
  const int L  = width ? W : H;
  const long n = (long)B * C * H * W;
  float* out = ws.alloc(n);
  size_t mark = ws.used;
  float* xp  = ws.alloc((long)NB * C * L);
  float* qkv = ws.alloc((long)NB * 192 * L);
  float* sv  = ws.alloc((long)NB * C * L);

  if (width)   // (B,C,H,W) -> (B,H,C,W)
    k_permute4<<<nblk(n, 256), 256, 0, s>>>(x, xp, n, H, C, W,
        (long)C * H * W, (long)W, (long)H * W, 1L);
  else         // (B,C,H,W) -> (B,W,C,H)
    k_permute4<<<nblk(n, 256), 256, 0, s>>>(x, xp, n, W, C, H,
        (long)C * H * W, 1L, (long)H * W, (long)W);

  gemm(ws, s, qkv_w, xp, nullptr, qkv, 192, 64, L, NB);
  bn(s, qkv, bqg, bqb, mean, istd, 192, NB, L, 0);

  const long rows = (long)NB * 8 * L;
  k_zero<<<1, 32, 0, s>>>(asums, 16);
  k_attn_stats<<<nblk(rows, 256), 256, 0, s>>>(qkv, asums, NB, L);
  k_stats_finalize<<<1, 32, 0, s>>>(asums, mean, istd, 8,
                                    (float)((double)NB * L * L));
  k_attn_apply<<<nblk(rows, 8), 256, 0, s>>>(qkv, mean, istd, bsg, bsb, sv, NB, L);
  bn(s, sv, bog, bob, mean, istd, 64, NB, L, 0);

  if (width)   // (B*H,64,W) -> (B,64,H,W)
    k_permute4<<<nblk(n, 256), 256, 0, s>>>(sv, out, n, C, H, W,
        (long)H * C * W, (long)W, (long)C * W, 1L);
  else         // (B*W,64,H) -> (B,64,H,W)
    k_permute4<<<nblk(n, 256), 256, 0, s>>>(sv, out, n, C, H, W,
        (long)W * C * H, (long)H, 1L, (long)C * H);
  ws.used = mark;
  return out;
}

}  // namespace

extern "C" void kernel_launch(void* const* d_in, const int* in_sizes, int n_in,
                              void* d_out, int out_size, void* d_ws, size_t ws_size,
                              hipStream_t stream) {
  (void)in_sizes; (void)n_in; (void)out_size;
  Cur P{d_in, 0};
  const float* x = P.next();               // (8, 342, 512)
  Ws ws{(char*)d_ws, 0, ws_size};
  float* mean  = ws.alloc(1024);
  float* istd  = ws.alloc(1024);
  float* asums = ws.alloc(32);

  const int B = 8, L1 = 512, H = 272;

  // ---- TCN stack ----
  float* o = tcn_block(ws, P, stream, mean, istd, x, B, 342, 342, L1, 1);
  o = tcn_block(ws, P, stream, mean, istd, o, B, 342, 306, L1, 2);
  o = tcn_block(ws, P, stream, mean, istd, o, B, 306, 288, L1, 4);

  // ---- projection 288 -> 272 (becomes the 2D input (B,1,272,512)) ----
  const float* pw = P.next(); const float* pg = P.next(); const float* pb = P.next();
  float* proj = ws.alloc((long)B * 272 * L1);
  gemm(ws, stream, pw, o, nullptr, proj, 272, 288, L1, B);
  bn(stream, proj, pg, pb, mean, istd, 272, B, L1, 1);

  // ---- conv2d residual stack (stride halves W each res block) ----
  o = conv_block(ws, P, stream, mean, istd, proj, B, 1, 8, H, 512, 1);  // up
  o = conv_block(ws, P, stream, mean, istd, o, B, 8,  8,  H, 512, 2);   // W=256
  o = conv_block(ws, P, stream, mean, istd, o, B, 8,  16, H, 256, 2);   // W=128
  o = conv_block(ws, P, stream, mean, istd, o, B, 16, 32, H, 128, 2);   // W=64
  o = conv_block(ws, P, stream, mean, istd, o, B, 32, 64, H, 64,  2);   // W=32

  // ---- axial attention: width then height ----
  o = axial(ws, P, stream, mean, istd, asums, o, B, H, 32, true);
  o = axial(ws, P, stream, mean, istd, asums, o, B, H, 32, false);

  // ---- final head: 1x1 convs as GEMM over S = H*W ----
  const float* c1w = P.next(); const float* c1b = P.next();
  const float* fg  = P.next(); const float* fb  = P.next();
  const float* c2w = P.next(); const float* c2b = P.next();
  const long S = (long)H * 32;                      // 8704 (multiple of 16)
  float* f1 = ws.alloc((long)B * 32 * S);
  gemm(ws, stream, c1w, o, c1b, f1, 32, 64, (int)S, B);
  bn(stream, f1, fg, fb, mean, istd, 32, B, S, 1);
  gemm(ws, stream, c2w, f1, c2b, (float*)d_out, 3, 32, (int)S, B);
}
